// Predictor_23562190586222
// MI455X (gfx1250) — matrix operand order repaired and checked by reading, not compiled, not run on hardware
//
#include <hip/hip_runtime.h>
#include <hip/hip_bf16.h>

typedef __attribute__((ext_vector_type(16))) __bf16 v16bf;
typedef __attribute__((ext_vector_type(8)))  float  v8f;

__device__ __forceinline__ unsigned short f2bf(float f) {
    unsigned u = __float_as_uint(f);
    unsigned r = (u + 0x7FFFu + ((u >> 16) & 1u)) >> 16;
    return (unsigned short)r;
}
__device__ __forceinline__ float bf2f(unsigned short h) {
    return __uint_as_float((unsigned)h << 16);
}

union Frag { v16bf v; unsigned u[8]; uint4 q[2]; };

// A fragment: 16-bit 16x32 A layout. Lane covers row (lane&15); VGPR v holds
// K = (v<4?0:16) + 8*half + 2*(v&3)  ->  q[0] = channels [c0+8h, +8),
//                                        q[1] = channels [c0+16+8h, +8).
template<int CIN>
__device__ __forceinline__ void load_a(Frag& A, const unsigned short* __restrict__ xrow,
                                       int c0, int half) {
    const uint4 z4 = {0u, 0u, 0u, 0u};
    if constexpr (CIN >= 32) {
        A.q[0] = *(const uint4*)(xrow + c0 + 8 * half);
        A.q[1] = *(const uint4*)(xrow + c0 + 16 + 8 * half);
    } else if constexpr (CIN == 16) {
        A.q[0] = *(const uint4*)(xrow + 8 * half);
        A.q[1] = z4;
    } else if constexpr (CIN == 8) {
        A.q[0] = (half == 0) ? *(const uint4*)(xrow) : z4;
        A.q[1] = z4;
    } else { // CIN == 4
        A.q[0] = z4; A.q[1] = z4;
        if (half == 0) { uint2 t = *(const uint2*)(xrow); A.u[0] = t.x; A.u[1] = t.y; }
    }
}

// B fragment: 16-bit 32x16 B layout. Lane covers column (lane&15); VGPR v holds
// K = (v<4?0:16) + 8*half + 2*(v&3)  ->  q[0] = channels [c0+8h, +8), q[1] = [c0+16+8h, +8).
template<int CIN>
__device__ __forceinline__ void load_b(Frag& B, const unsigned short* __restrict__ wrow,
                                       int c0, int half) {
    const uint4 z4 = {0u, 0u, 0u, 0u};
    if constexpr (CIN >= 32) {
        B.q[0] = *(const uint4*)(wrow + c0 + 8 * half);
        B.q[1] = *(const uint4*)(wrow + c0 + 16 + 8 * half);
    } else if constexpr (CIN == 16) {
        B.q[0] = *(const uint4*)(wrow + 8 * half);
        B.q[1] = z4;
    } else if constexpr (CIN == 8) {
        B.q[0] = (half == 0) ? *(const uint4*)(wrow) : z4;
        B.q[1] = z4;
    } else { // CIN == 4
        B.q[0] = z4; B.q[1] = z4;
        if (half == 0) { uint2 t = *(const uint2*)(wrow); B.u[0] = t.x; B.u[1] = t.y; }
    }
}

// ---------------------------------------------------------------------------
// Gather + WMMA sconv, register-blocked over N, with optional fused residual.
// One wave computes a 16(M) x 16*NT(N) strip: A is gathered ONCE per (k, c0)
// and reused for NT back-to-back v_wmma_f32_16x16x32_bf16.
// resid (nullable): bf16 [*, ldo] read at the same offsets as the store and
// added before writeout (enables in-place inception residual).
// ---------------------------------------------------------------------------
template<int CIN, int NT>
__global__ __launch_bounds__(256) void sconv_wmma_t(
    const unsigned short* __restrict__ x,
    const int* __restrict__ nbr, int K, int has_nbr,
    const unsigned short* __restrict__ wt,     // [K][Cout][CIN] bf16
    const float* __restrict__ bias,
    unsigned short* __restrict__ outb, int ldo, int coloff,
    float* __restrict__ outf,
    const unsigned short* __restrict__ resid,
    int Nout, int Cout, int do_relu)
{
    const int tm = blockIdx.x * 8 + (threadIdx.x >> 5);
    if (tm * 16 >= Nout) return;              // wave-uniform: EXEC all-ones inside
    const int lane = threadIdx.x & 31;
    const int half = lane >> 4;
    const int mr   = lane & 15;
    const int row  = tm * 16 + mr;            // A row this lane gathers
    const bool rowok = row < Nout;
    const uint4 z4 = {0u, 0u, 0u, 0u};

    bool colok[NT];
    const unsigned short* wcol[NT];
#pragma unroll
    for (int t = 0; t < NT; ++t) {
        int ncol = t * 16 + mr;
        colok[t] = ncol < Cout;
        wcol[t]  = wt + (size_t)(colok[t] ? ncol : 0) * CIN;
    }
    const size_t wks = (size_t)Cout * CIN;    // weight stride per kernel offset

    const int* nrow = (has_nbr && rowok) ? (nbr + (size_t)row * K) : nullptr;

    constexpr int NCH = (CIN + 31) / 32;
    v8f zero8 = {};
    v8f acc[NT];
#pragma unroll
    for (int t = 0; t < NT; ++t) acc[t] = zero8;

    for (int k = 0; k < K; ++k) {
        int ridx = -1;
        if (rowok) ridx = nrow ? nrow[k] : row;
        const unsigned short* xrow = x + (size_t)(ridx < 0 ? 0 : ridx) * CIN;
#pragma unroll
        for (int ch = 0; ch < NCH; ++ch) {
            const int c0 = ch * 32;
            Frag A;
            if (ridx >= 0) load_a<CIN>(A, xrow, c0, half);
            else           { A.q[0] = z4; A.q[1] = z4; }
#pragma unroll
            for (int t = 0; t < NT; ++t) {
                Frag B;
                if (colok[t]) load_b<CIN>(B, wcol[t] + k * wks, c0, half);
                else          { B.q[0] = z4; B.q[1] = z4; }
                acc[t] = __builtin_amdgcn_wmma_f32_16x16x32_bf16(
                    false, A.v, false, B.v, (short)0, acc[t], false, false);
            }
        }
    }

#pragma unroll
    for (int t = 0; t < NT; ++t) {
        int ncol = t * 16 + mr;
        float bv = (bias && colok[t]) ? bias[ncol] : 0.0f;
#pragma unroll
        for (int r = 0; r < 8; ++r) {
            int orow = tm * 16 + r + 8 * half;  // C/D layout: VGPR r -> M = r + 8*half
            if (orow >= Nout || !colok[t]) continue;
            size_t off = (size_t)orow * ldo + coloff + ncol;
            float val = acc[t][r] + bv;
            if (resid)   val += bf2f(resid[off]);
            if (do_relu) val = fmaxf(val, 0.0f);
            if (outf) outf[off] = val;
            else      outb[off] = f2bf(val);
        }
    }
}

// w [K][Cin][Cout] f32  ->  wt [K][Cout][Cin] bf16
__global__ void prep_w_kernel(const float* __restrict__ w,
                              unsigned short* __restrict__ wt,
                              int K, int Cin, int Cout)
{
    int i = blockIdx.x * blockDim.x + threadIdx.x;
    int total = K * Cin * Cout;
    if (i >= total) return;
    int d = i % Cout; int t = i / Cout; int c = t % Cin; int k = t / Cin;
    wt[((size_t)k * Cout + d) * Cin + c] = f2bf(w[i]);
}

// f32 [rows][cols] -> bf16 at dst[:, coloff:coloff+cols], row stride ldo
__global__ void cvt_cols_kernel(const float* __restrict__ src,
                                unsigned short* __restrict__ dst,
                                int rows, int cols, int ldo, int coloff)
{
    int i = blockIdx.x * blockDim.x + threadIdx.x;
    if (i >= rows * cols) return;
    int r = i / cols, c = i % cols;
    dst[(size_t)r * ldo + coloff + c] = f2bf(src[i]);
}

// ---------------------------------------------------------------------------
extern "C" void kernel_launch(void* const* d_in, const int* in_sizes, int n_in,
                              void* d_out, int out_size, void* d_ws, size_t ws_size,
                              hipStream_t stream)
{
    (void)in_sizes; (void)n_in; (void)out_size; (void)ws_size;
    const int N0 = 300000, N1 = 100000, N2 = 40000, N3 = 18000, NT_ = 22000;
    const int K3 = 27, K2 = 8;

    const float* x0 = (const float*)d_in[0];   // [N3,8]
    const float* x1 = (const float*)d_in[1];   // [N3,32]
    const float* x2 = (const float*)d_in[2];   // [N2,64]
    const float* x3 = (const float*)d_in[3];   // [N1,32]
    const float* x4 = (const float*)d_in[4];   // [N0,16]
    const int* nbr0  = (const int*)d_in[5];
    const int* nbrd0 = (const int*)d_in[6];
    const int* nbr1  = (const int*)d_in[7];
    const int* nbrd1 = (const int*)d_in[8];
    const int* nbr2  = (const int*)d_in[9];
    const int* nbrd2 = (const int*)d_in[10];
    const int* nbr3  = (const int*)d_in[11];
    const int* nbrt  = (const int*)d_in[12];
    const int* nbrts = (const int*)d_in[13];

    struct Conv { const float* w; const float* b; int K, Cin, Cout; unsigned short* wt; };
    Conv convs[70]; int nc = 0; int pi = 14;
    auto addconv = [&](int K, int Cin, int Cout) {
        convs[nc].w = (const float*)d_in[pi++];
        convs[nc].b = (const float*)d_in[pi++];
        convs[nc].K = K; convs[nc].Cin = Cin; convs[nc].Cout = Cout;
        convs[nc].wt = nullptr; nc++;
    };
    auto addblocks = [&](int ch) {
        int c4 = ch / 4, c2 = ch / 2;
        for (int r = 0; r < 3; ++r) {
            addconv(K3, ch, c4);  // c00
            addconv(K3, c4, c2);  // c01
            addconv(1,  ch, c4);  // c10 (1x1)
            addconv(K3, c4, c4);  // c11
            addconv(1,  c4, c2);  // c12 (1x1)
        }
    };
    addconv(K3, 16, 16);        // 0  conv0
    addconv(K2, 16, 32);        // 1  down0
    addblocks(64);              // 2..16  block0
    addconv(K3, 64, 32);        // 17 conv1
    addconv(K2, 32, 64);        // 18 down1
    addblocks(128);             // 19..33 block1
    addconv(K3, 128, 64);       // 34 conv2
    addconv(K2, 64, 32);        // 35 down2
    addblocks(64);              // 36..50 block2
    addconv(K3, 64, 8);         // 51 conv3
    addconv(K3, 16, 16);        // 52 conv_coord
    addblocks(16);              // 53..67 block3
    addconv(K3, 16, 8);         // 68 conv4

    char* wsp = (char*)d_ws; size_t used = 0;
    auto alloc = [&](size_t bytes) -> void* {
        size_t a = (used + 255) & ~(size_t)255;
        void* p = wsp + a; used = a + bytes; return p;
    };

    for (int i = 0; i < nc; ++i) {
        size_t e = (size_t)convs[i].K * convs[i].Cin * convs[i].Cout;
        convs[i].wt = (unsigned short*)alloc(e * 2);
        prep_w_kernel<<<(int)((e + 255) / 256), 256, 0, stream>>>(
            convs[i].w, convs[i].wt, convs[i].K, convs[i].Cin, convs[i].Cout);
    }

    auto abuf = [&](size_t elems) { return (unsigned short*)alloc(elems * 2); };
    unsigned short* xb4 = abuf((size_t)N0 * 16);
    unsigned short* t0  = abuf((size_t)N0 * 16);
    unsigned short* t1  = abuf((size_t)N1 * 64);
    unsigned short* t2  = abuf((size_t)N2 * 128);
    unsigned short* t3  = abuf((size_t)N3 * 64);
    unsigned short* t4  = abuf((size_t)N3 * 16);
    unsigned short* t5  = abuf((size_t)NT_ * 16);
    unsigned short* sA  = abuf((size_t)N1 * 32);
    unsigned short* sB  = abuf((size_t)N1 * 16);

    auto launch_sconv = [&](const unsigned short* x, const int* nbr, int K,
                            const Conv& cv, unsigned short* outb, float* outf,
                            int ldo, int coloff, const unsigned short* resid,
                            int Nout, int relu) {
        int tiles_m = (Nout + 15) / 16;
        int nt = (cv.Cout + 15) / 16;         // 1, 2 or 4 in this network
        dim3 grid((tiles_m + 7) / 8);
        int hn = nbr != nullptr;
#define DISP(CI, NTV) sconv_wmma_t<CI, NTV><<<grid, 256, 0, stream>>>( \
            x, nbr, K, hn, cv.wt, cv.b, outb, ldo, coloff, outf, resid, Nout, cv.Cout, relu)
        switch (cv.Cin) {
        case 4:   DISP(4, 1);  break;                          // (4,4) (4,8)
        case 16:  if (nt <= 1) DISP(16, 1); else DISP(16, 2);  break;
        case 32:  if (nt <= 2) DISP(32, 2); else DISP(32, 4);  break;
        case 64:  if (nt <= 1) DISP(64, 1); else DISP(64, 2);  break;
        default:  if (nt <= 2) DISP(128, 2); else DISP(128, 4); break;
        }
#undef DISP
    };
    auto cvt = [&](const float* src, unsigned short* dst, int rows, int cols,
                   int ldo, int coloff) {
        size_t n = (size_t)rows * cols;
        cvt_cols_kernel<<<(int)((n + 255) / 256), 256, 0, stream>>>(
            src, dst, rows, cols, ldo, coloff);
    };
    // Inception-ResNet block, residual fused into the two column-strip stores:
    //   sA = relu(sconv(t, c00));  sB = relu(t @ c10)   (both readers of t first)
    //   t[:, 0:c2)  = sconv(sA, c01) + t[:, 0:c2)       (in-place, disjoint cols)
    //   sA = relu(sconv(sB, c11))
    //   t[:, c2:ch) = (sA @ c12)   + t[:, c2:ch)
    auto run_block3 = [&](unsigned short* t, int N, int ch,
                          const int* nbr, int ci0) {
        int c4 = ch / 4, c2 = ch / 2;
        for (int r = 0; r < 3; ++r) {
            const Conv& c00 = convs[ci0 + r * 5 + 0];
            const Conv& c01 = convs[ci0 + r * 5 + 1];
            const Conv& c10 = convs[ci0 + r * 5 + 2];
            const Conv& c11 = convs[ci0 + r * 5 + 3];
            const Conv& c12 = convs[ci0 + r * 5 + 4];
            launch_sconv(t,  nbr, K3, c00, sA, nullptr, c4, 0, nullptr, N, 1);
            launch_sconv(t,  nullptr, 1, c10, sB, nullptr, c4, 0, nullptr, N, 1);
            launch_sconv(sA, nbr, K3, c01, t, nullptr, ch, 0,  t, N, 0);
            launch_sconv(sB, nbr, K3, c11, sA, nullptr, c4, 0, nullptr, N, 1);
            launch_sconv(sA, nullptr, 1, c12, t, nullptr, ch, c2, t, N, 0);
        }
    };

    // ---- network ----
    cvt(x4, xb4, N0, 16, 16, 0);
    launch_sconv(xb4, nbr0,  K3, convs[0],  t0, nullptr, 16, 0, nullptr, N0, 1); // conv0
    launch_sconv(t0,  nbrd0, K2, convs[1],  t1, nullptr, 64, 0, nullptr, N1, 1); // down0
    cvt(x3, t1, N1, 32, 64, 32);
    run_block3(t1, N1, 64, nbr1, 2);                                              // block0
    launch_sconv(t1,  nbr1,  K3, convs[17], sA, nullptr, 32, 0, nullptr, N1, 1); // conv1
    launch_sconv(sA,  nbrd1, K2, convs[18], t2, nullptr, 128, 0, nullptr, N2, 1);// down1
    cvt(x2, t2, N2, 64, 128, 64);
    run_block3(t2, N2, 128, nbr2, 19);                                            // block1
    launch_sconv(t2,  nbr2,  K3, convs[34], sA, nullptr, 64, 0, nullptr, N2, 1); // conv2
    launch_sconv(sA,  nbrd2, K2, convs[35], t3, nullptr, 64, 0, nullptr, N3, 1); // down2
    cvt(x1, t3, N3, 32, 64, 32);
    run_block3(t3, N3, 64, nbr3, 36);                                             // block2
    launch_sconv(t3,  nbr3,  K3, convs[51], t4, nullptr, 16, 0, nullptr, N3, 0); // conv3
    cvt(x0, t4, N3, 8, 16, 8);
    launch_sconv(t4,  nbrt,  K3, convs[52], t5, nullptr, 16, 0, nullptr, NT_, 1);// conv_coord
    run_block3(t5, NT_, 16, nbrts, 53);                                           // block3
    launch_sconv(t5,  nbrts, K3, convs[68], nullptr, (float*)d_out, 8, 0, nullptr, NT_, 0); // conv4
}
